// S2ANetHead_24799141167131
// MI455X (gfx1250) — matrix operand order repaired and checked by reading, not compiled, not run on hardware
//
#include <hip/hip_runtime.h>
#include <hip/hip_bf16.h>

#define HH 128
#define WW 128
#define HW (HH * WW)
#define CC 256
#define NCLS 15

typedef __attribute__((ext_vector_type(16))) __bf16 v16bf;
typedef __attribute__((ext_vector_type(8)))  float  v8f;
typedef __attribute__((ext_vector_type(8)))  unsigned short v8u;

union V16U { v8u h[2]; v16bf bf; };

// float -> bf16 bits, round-to-nearest-even
__device__ inline unsigned short f2bf_bits(float f) {
    unsigned int x = __float_as_uint(f);
    unsigned int lsb = (x >> 16) & 1u;
    x += 0x7fffu + lsb;
    return (unsigned short)(x >> 16);
}

__device__ inline float bilinear_tap(const float* __restrict__ xc, float ys, float xs) {
    float x0f = floorf(xs), y0f = floorf(ys);
    float wx1 = xs - x0f, wx0 = 1.f - wx1;
    float wy1 = ys - y0f, wy0 = 1.f - wy1;
    int x0 = (int)x0f, y0 = (int)y0f;
    float acc = 0.f;
#pragma unroll
    for (int t = 0; t < 4; ++t) {
        int yy = y0 + (t >> 1);
        int xx = x0 + (t & 1);
        float w = ((t >> 1) ? wy1 : wy0) * ((t & 1) ? wx1 : wx0);
        if ((unsigned)yy < (unsigned)HH && (unsigned)xx < (unsigned)WW)
            acc += xc[yy * WW + xx] * w;
    }
    return acc;
}

// -------------------------------------------------------------------------
// Stage one K-step (32 k-values) of A (bf16 weights, contiguous) and
// B (implicit im2col, fp32->bf16) into LDS.
// B mapping: thread = column n (fully coalesced global loads, since an
// N-tile of 128 == one image row -> py is wave-uniform), j-loop walks 16
// contiguous k slots -> 2x ds_store_b128 per thread.
// -------------------------------------------------------------------------
template <int SAMPLED>
__device__ inline void stage_tiles(unsigned short (* __restrict__ As)[40],
                                   unsigned short (* __restrict__ Bs)[40],
                                   const float* __restrict__ X,
                                   const unsigned short* __restrict__ Wb,
                                   int Ktot, int co0, int p0, int k0, int tid,
                                   const float* __restrict__ xsbuf,
                                   const float* __restrict__ ysbuf) {
    // ---- A tile: 128 rows x 32 k of bf16 weights ----
    {
        const int lrow = tid >> 1;
        const int lkh  = (tid & 1) << 4;
        const unsigned short* wsrc = Wb + (size_t)(co0 + lrow) * Ktot + k0 + lkh;
        __builtin_prefetch(wsrc + 32, 0, 1);
        v8u a0 = *(const v8u*)wsrc;
        v8u a1 = *(const v8u*)(wsrc + 8);
        *(v8u*)&As[lrow][lkh]     = a0;
        *(v8u*)&As[lrow][lkh + 8] = a1;
    }
    // ---- B tile: 128 cols x 32 k, row-coalesced gather ----
    {
        const int n  = tid & 127;
        const int kh = (tid >> 7) << 4;   // 0 / 16
        const int gp = p0 + n;
        const int py = gp >> 7;           // wave-uniform (p0 = 128*bx)
        const int px = gp & (WW - 1);
        alignas(16) unsigned short tmp[16];
#pragma unroll
        for (int j = 0; j < 16; ++j) {
            int k  = k0 + kh + j;         // uniform across the 128 threads of this half
            int ci = k / 9;
            int r9 = k - ci * 9;
            float v;
            if (SAMPLED) {
                float xs = xsbuf[r9 * HW + gp];
                float ys = ysbuf[r9 * HW + gp];
                v = bilinear_tap(X + (size_t)ci * HW, ys, xs);
            } else {
                int dy = r9 / 3;
                int dx = r9 - dy * 3;
                int yy = py + dy - 1;
                int xx = px + dx - 1;
                v = ((unsigned)yy < (unsigned)HH && (unsigned)xx < (unsigned)WW)
                        ? X[(size_t)ci * HW + yy * WW + xx] : 0.f;
            }
            tmp[j] = f2bf_bits(v);
        }
        *(v8u*)&Bs[n][kh]     = *(const v8u*)&tmp[0];
        *(v8u*)&Bs[n][kh + 8] = *(const v8u*)&tmp[8];
    }
}

// fragments in contiguous k order (k = 16*(lane/16) + s) for A and B + 8 WMMAs
__device__ inline void mma_step(const unsigned short (* __restrict__ As)[40],
                                const unsigned short (* __restrict__ Bs)[40],
                                v8f acc[2][4], int lane, int mw, int nw) {
    const int am  = lane & 15;
    const int kh8 = (lane >> 4) << 4;   // A: lanes16-31 hold K 16..31
    V16U afrag[2];
#pragma unroll
    for (int f = 0; f < 2; ++f) {
        const unsigned short* ap = &As[mw * 32 + f * 16 + am][0];
        afrag[f].h[0] = *(const v8u*)(ap + kh8);
        afrag[f].h[1] = *(const v8u*)(ap + 8 + kh8);
    }
    V16U bfrag[4];
    const int bk = (lane >> 4) << 4;    // B: lanes16-31 hold K 16..31
#pragma unroll
    for (int g = 0; g < 4; ++g) {
        const unsigned short* bp = &Bs[nw * 64 + g * 16 + am][bk];
        bfrag[g].h[0] = *(const v8u*)bp;
        bfrag[g].h[1] = *(const v8u*)(bp + 8);
    }
#pragma unroll
    for (int f = 0; f < 2; ++f)
#pragma unroll
        for (int g = 0; g < 4; ++g)
            acc[f][g] = __builtin_amdgcn_wmma_f32_16x16x32_bf16(
                false, afrag[f].bf, false, bfrag[g].bf,
                (short)0, acc[f][g], false, false);
}

// -------------------------------------------------------------------------
// Implicit-GEMM 3x3 conv, double-buffered LDS (one barrier per K-step):
// M=Co tile 128 (blockIdx.y), N=HW tile 128 (blockIdx.x), K=Cin*9 step 32.
// 8 wave32s: 4 over M, 2 over N -> 2x4 wmma frags per wave.
// -------------------------------------------------------------------------
template <int SAMPLED>
__global__ __launch_bounds__(256)
void conv_gemm_kernel(const float* __restrict__ X,
                      const unsigned short* __restrict__ Wb,   // bf16 bits [Co][Cin*9]
                      const float* __restrict__ bias,
                      float* __restrict__ Y,
                      int Cin, int relu,
                      const float* __restrict__ xsbuf,
                      const float* __restrict__ ysbuf) {
    const int Ktot = Cin * 9;
    const int p0  = blockIdx.x * 128;
    const int co0 = blockIdx.y * 128;

    __shared__ alignas(16) unsigned short As[2][128][40];  // ping-pong, 80B rows
    __shared__ alignas(16) unsigned short Bs[2][128][40];

    const int tid  = threadIdx.x;
    const int lane = tid & 31;
    const int wave = tid >> 5;
    const int mw = wave & 3;   // 4 waves over M
    const int nw = wave >> 2;  // 2 waves over N

    const v8f vzero = {0.f, 0.f, 0.f, 0.f, 0.f, 0.f, 0.f, 0.f};
    v8f acc[2][4];
#pragma unroll
    for (int f = 0; f < 2; ++f)
#pragma unroll
        for (int g = 0; g < 4; ++g) acc[f][g] = vzero;

    int cur = 0;
    stage_tiles<SAMPLED>(As[0], Bs[0], X, Wb, Ktot, co0, p0, 0, tid, xsbuf, ysbuf);
    __syncthreads();
    for (int k0 = 32; k0 < Ktot; k0 += 32) {
        stage_tiles<SAMPLED>(As[cur ^ 1], Bs[cur ^ 1], X, Wb, Ktot, co0, p0, k0, tid, xsbuf, ysbuf);
        mma_step(As[cur], Bs[cur], acc, lane, mw, nw);
        __syncthreads();
        cur ^= 1;
    }
    mma_step(As[cur], Bs[cur], acc, lane, mw, nw);

    // ---- epilogue: C/D layout: vgpr e -> M = (lane>=16?8:0)+e, N = lane&15 ----
    const int nlo = lane & 15;
    const int m8  = (lane >> 4) << 3;
#pragma unroll
    for (int f = 0; f < 2; ++f)
#pragma unroll
        for (int g = 0; g < 4; ++g)
#pragma unroll
            for (int e = 0; e < 8; ++e) {
                int co = co0 + mw * 32 + f * 16 + m8 + e;
                int p  = p0 + nw * 64 + g * 16 + nlo;
                float v = acc[f][g][e] + bias[co];
                if (relu) v = fmaxf(v, 0.f);
                Y[(size_t)co * HW + p] = v;
            }
}

// -------------------------------------------------------------------------
// small heads (Co = 5/15): direct conv, ksize 1 or 3 (pad = ksize/2)
// -------------------------------------------------------------------------
__global__ void head_conv_kernel(const float* __restrict__ X, const float* __restrict__ Wt,
                                 const float* __restrict__ b, float* __restrict__ Y,
                                 int Cin, int Co, int ksize) {
    int idx = blockIdx.x * blockDim.x + threadIdx.x;
    if (idx >= Co * HW) return;
    int co = idx / HW;
    int p  = idx - co * HW;
    int py = p >> 7, px = p & (WW - 1);
    float acc = b[co];
    if (ksize == 1) {
        const float* w = Wt + (size_t)co * Cin;
        for (int ci = 0; ci < Cin; ++ci)
            acc += X[(size_t)ci * HW + p] * w[ci];
    } else {
        const float* w = Wt + (size_t)co * Cin * 9;
        for (int ci = 0; ci < Cin; ++ci) {
            const float* xc = X + (size_t)ci * HW;
            const float* wc = w + ci * 9;
#pragma unroll
            for (int kk = 0; kk < 9; ++kk) {
                int yy = py + (kk / 3) - 1;
                int xx = px + (kk % 3) - 1;
                if ((unsigned)yy < (unsigned)HH && (unsigned)xx < (unsigned)WW)
                    acc += xc[yy * WW + xx] * wc[kk];
            }
        }
    }
    Y[idx] = acc;
}

// fp32 -> bf16-bits flat convert (weight prepack)
__global__ void cvt_bf16_kernel(const float* __restrict__ in, unsigned short* __restrict__ out, int n) {
    int i = blockIdx.x * blockDim.x + threadIdx.x;
    if (i < n) out[i] = f2bf_bits(in[i]);
}

// ORConv weight rotation: wr[o*8+r][ci*9+kk] = or_w[o][ci][src_r(kk)], br[o*8+r]=or_b[o]
__device__ __constant__ int RINGd[8] = {0, 1, 2, 5, 8, 7, 6, 3};
__global__ void rotate_orw_kernel(const float* __restrict__ orw, const float* __restrict__ orb,
                                  unsigned short* __restrict__ wr, float* __restrict__ br) {
    int idx = blockIdx.x * blockDim.x + threadIdx.x;  // over 256*256 (c, ci)
    if (idx >= 256 * 256) return;
    int c  = idx >> 8;
    int ci = idx & 255;
    int o = c >> 3, r = c & 7;
    int src[9];
#pragma unroll
    for (int j = 0; j < 9; ++j) src[j] = j;
#pragma unroll
    for (int j = 0; j < 8; ++j) src[RINGd[j]] = RINGd[(j - r) & 7];
    const float* w = orw + ((size_t)o * 256 + ci) * 9;
    unsigned short* d = wr + (size_t)c * (256 * 9) + ci * 9;
#pragma unroll
    for (int kk = 0; kk < 9; ++kk) d[kk] = f2bf_bits(w[src[kk]]);
    if (ci == 0) br[c] = orb[o];
}

// anchors + rbox decode -> anchors[HW,5], refine[HW,5]
__global__ void decode_kernel(const float* __restrict__ pred,  // [5][HW]
                              float* __restrict__ anchors, float* __restrict__ refine,
                              const int* __restrict__ stride_p) {
    int p = blockIdx.x * blockDim.x + threadIdx.x;
    if (p >= HW) return;
    const float PI = 3.14159265358979323846f;
    float s = (float)(*stride_p);
    int py = p >> 7, px = p & (WW - 1);
    float ctr = 0.5f * (8.0f - 1.0f);
    float ax = px * s + ctr, ay = py * s + ctr;
    const float aw = 32.f, ah = 32.f, aa = 0.f;
    anchors[p * 5 + 0] = ax; anchors[p * 5 + 1] = ay;
    anchors[p * 5 + 2] = aw; anchors[p * 5 + 3] = ah;
    anchors[p * 5 + 4] = aa;
    float dx = pred[0 * HW + p], dy = pred[1 * HW + p];
    float dw = pred[2 * HW + p], dh = pred[3 * HW + p], dt = pred[4 * HW + p];
    const float maxr = 13.815510558f;  // |log(1e-6)|
    dw = fminf(fmaxf(dw, -maxr), maxr);
    dh = fminf(fmaxf(dh, -maxr), maxr);
    float cs = cosf(aa), sn = sinf(aa);
    float gx = dx * aw * cs - dy * ah * sn + ax;
    float gy = dx * aw * sn + dy * ah * cs + ay;
    float gw = aw * expf(dw);
    float gh = ah * expf(dh);
    float a = aa + dt + PI * 0.25f;
    float t = fmodf(a, PI);
    if (t < 0.f) t += PI;
    float ga = t - PI * 0.25f;
    refine[p * 5 + 0] = gx; refine[p * 5 + 1] = gy;
    refine[p * 5 + 2] = gw; refine[p * 5 + 3] = gh;
    refine[p * 5 + 4] = ga;
}

// per-pixel sampling grid for the align (deformable) conv: xs/ys [9][HW]
__global__ void coords_kernel(const float* __restrict__ refine,
                              float* __restrict__ xs, float* __restrict__ ys,
                              const int* __restrict__ stride_p) {
    int p = blockIdx.x * blockDim.x + threadIdx.x;
    if (p >= HW) return;
    float s = (float)(*stride_p);
    float axc = refine[p * 5 + 0] / s;
    float ayc = refine[p * 5 + 1] / s;
    float aw  = refine[p * 5 + 2] / s;
    float ah  = refine[p * 5 + 3] / s;
    float aa  = refine[p * 5 + 4];
    float cs = cosf(aa), sn = sinf(aa);
#pragma unroll
    for (int kk = 0; kk < 9; ++kk) {
        float xxk = (float)((kk % 3) - 1);
        float yyk = (float)((kk / 3) - 1);
        float dx = (aw * (1.f / 3.f)) * xxk;
        float dy = (ah * (1.f / 3.f)) * yyk;
        xs[kk * HW + p] = cs * dx - sn * dy + axc;
        ys[kk * HW + p] = sn * dx + cs * dy + ayc;
    }
}

// rotation-invariant pooling: pooled[o][p] = max_r orf[o*8+r][p]
__global__ void rot_pool_kernel(const float* __restrict__ orf, float* __restrict__ pooled) {
    int idx = blockIdx.x * blockDim.x + threadIdx.x;
    if (idx >= 32 * HW) return;
    int o = idx / HW;
    int p = idx - o * HW;
    float m = orf[(size_t)(o * 8) * HW + p];
#pragma unroll
    for (int r = 1; r < 8; ++r)
        m = fmaxf(m, orf[(size_t)(o * 8 + r) * HW + p]);
    pooled[idx] = m;
}

// -------------------------------------------------------------------------
extern "C" void kernel_launch(void* const* d_in, const int* in_sizes, int n_in,
                              void* d_out, int out_size, void* d_ws, size_t ws_size,
                              hipStream_t stream) {
    (void)in_sizes; (void)n_in; (void)out_size; (void)ws_size;
    const float* feats      = (const float*)d_in[0];
    const float* fr_w0 = (const float*)d_in[1];  const float* fr_b0 = (const float*)d_in[2];
    const float* fr_w1 = (const float*)d_in[3];  const float* fr_b1 = (const float*)d_in[4];
    const float* fr_hw = (const float*)d_in[5];  const float* fr_hb = (const float*)d_in[6];
    const float* fc_w0 = (const float*)d_in[7];  const float* fc_b0 = (const float*)d_in[8];
    const float* fc_w1 = (const float*)d_in[9];  const float* fc_b1 = (const float*)d_in[10];
    const float* fc_hw = (const float*)d_in[11]; const float* fc_hb = (const float*)d_in[12];
    const float* al_w  = (const float*)d_in[13]; const float* al_b  = (const float*)d_in[14];
    const float* or_w  = (const float*)d_in[15]; const float* or_b  = (const float*)d_in[16];
    const float* dr_w0 = (const float*)d_in[17]; const float* dr_b0 = (const float*)d_in[18];
    const float* dr_w1 = (const float*)d_in[19]; const float* dr_b1 = (const float*)d_in[20];
    const float* dr_hw = (const float*)d_in[21]; const float* dr_hb = (const float*)d_in[22];
    const float* dc_w0 = (const float*)d_in[23]; const float* dc_b0 = (const float*)d_in[24];
    const float* dc_w1 = (const float*)d_in[25]; const float* dc_b1 = (const float*)d_in[26];
    const float* dc_hw = (const float*)d_in[27]; const float* dc_hb = (const float*)d_in[28];
    const int*   stride_p = (const int*)d_in[29];

    float* out = (float*)d_out;
    float* out_fam_cls  = out + 0;                 // [15][HW]
    float* out_fam_bbox = out + (size_t)NCLS * HW; // [5][HW]
    float* out_odm_cls  = out_fam_bbox + (size_t)5 * HW;
    float* out_odm_bbox = out_odm_cls  + (size_t)NCLS * HW;
    float* out_anchors  = out_odm_bbox + (size_t)5 * HW;  // [HW,5]
    float* out_refine   = out_anchors  + (size_t)5 * HW;  // [HW,5]

    char* ws = (char*)d_ws;
    size_t off = 0;
    auto carve = [&](size_t bytes) -> void* {
        void* p = ws + off;
        off = (off + bytes + 255) & ~(size_t)255;
        return p;
    };
    const size_t ACT = (size_t)CC * HW * sizeof(float);
    float* bufA   = (float*)carve(ACT);
    float* bufB   = (float*)carve(ACT);
    float* orf    = (float*)carve(ACT);
    float* pooled = (float*)carve((size_t)32 * HW * sizeof(float));
    float* xsb    = (float*)carve((size_t)9 * HW * sizeof(float));
    float* ysb    = (float*)carve((size_t)9 * HW * sizeof(float));
    const size_t WB_FULL  = (size_t)CC * CC * 9 * sizeof(unsigned short);
    const size_t WB_SMALL = (size_t)CC * 32 * 9 * sizeof(unsigned short);
    unsigned short* wb_fr0  = (unsigned short*)carve(WB_FULL);
    unsigned short* wb_fr1  = (unsigned short*)carve(WB_FULL);
    unsigned short* wb_fc0  = (unsigned short*)carve(WB_FULL);
    unsigned short* wb_fc1  = (unsigned short*)carve(WB_FULL);
    unsigned short* wb_al   = (unsigned short*)carve(WB_FULL);
    unsigned short* wb_or   = (unsigned short*)carve(WB_FULL);
    unsigned short* wb_dr0  = (unsigned short*)carve(WB_FULL);
    unsigned short* wb_dr1  = (unsigned short*)carve(WB_FULL);
    unsigned short* wb_dc0  = (unsigned short*)carve(WB_SMALL);
    unsigned short* wb_dc1  = (unsigned short*)carve(WB_FULL);
    float* br_or = (float*)carve(256 * sizeof(float));

    const int NW_FULL  = CC * CC * 9;
    const int NW_SMALL = CC * 32 * 9;
    dim3 blk(256);
    auto gcvt = [&](int n) { return dim3((n + 255) / 256); };

    // ---- weight prepack (fp32 -> bf16 GEMM layout) ----
    cvt_bf16_kernel<<<gcvt(NW_FULL), blk, 0, stream>>>(fr_w0, wb_fr0, NW_FULL);
    cvt_bf16_kernel<<<gcvt(NW_FULL), blk, 0, stream>>>(fr_w1, wb_fr1, NW_FULL);
    cvt_bf16_kernel<<<gcvt(NW_FULL), blk, 0, stream>>>(fc_w0, wb_fc0, NW_FULL);
    cvt_bf16_kernel<<<gcvt(NW_FULL), blk, 0, stream>>>(fc_w1, wb_fc1, NW_FULL);
    cvt_bf16_kernel<<<gcvt(NW_FULL), blk, 0, stream>>>(al_w,  wb_al,  NW_FULL);
    cvt_bf16_kernel<<<gcvt(NW_FULL), blk, 0, stream>>>(dr_w0, wb_dr0, NW_FULL);
    cvt_bf16_kernel<<<gcvt(NW_FULL), blk, 0, stream>>>(dr_w1, wb_dr1, NW_FULL);
    cvt_bf16_kernel<<<gcvt(NW_SMALL), blk, 0, stream>>>(dc_w0, wb_dc0, NW_SMALL);
    cvt_bf16_kernel<<<gcvt(NW_FULL), blk, 0, stream>>>(dc_w1, wb_dc1, NW_FULL);
    rotate_orw_kernel<<<gcvt(256 * 256), blk, 0, stream>>>(or_w, or_b, wb_or, br_or);

    dim3 gemm_grid(HW / 128, CC / 128);  // (128, 2)

    // ---- FAM reg tower + 1x1 head -> fam_bbox_pred ----
    conv_gemm_kernel<0><<<gemm_grid, blk, 0, stream>>>(feats, wb_fr0, fr_b0, bufA, CC, 1, nullptr, nullptr);
    conv_gemm_kernel<0><<<gemm_grid, blk, 0, stream>>>(bufA,  wb_fr1, fr_b1, bufB, CC, 1, nullptr, nullptr);
    head_conv_kernel<<<gcvt(5 * HW), blk, 0, stream>>>(bufB, fr_hw, fr_hb, out_fam_bbox, CC, 5, 1);

    // ---- FAM cls tower + 1x1 head -> fam_cls_pred ----
    conv_gemm_kernel<0><<<gemm_grid, blk, 0, stream>>>(feats, wb_fc0, fc_b0, bufA, CC, 1, nullptr, nullptr);
    conv_gemm_kernel<0><<<gemm_grid, blk, 0, stream>>>(bufA,  wb_fc1, fc_b1, bufB, CC, 1, nullptr, nullptr);
    head_conv_kernel<<<gcvt(NCLS * HW), blk, 0, stream>>>(bufB, fc_hw, fc_hb, out_fam_cls, CC, NCLS, 1);

    // ---- anchors + decode + sampling grid ----
    decode_kernel<<<gcvt(HW), blk, 0, stream>>>(out_fam_bbox, out_anchors, out_refine, stride_p);
    coords_kernel<<<gcvt(HW), blk, 0, stream>>>(out_refine, xsb, ysb, stride_p);

    // ---- align conv (deformable, bilinear-sampled implicit GEMM) -> bufA ----
    conv_gemm_kernel<1><<<gemm_grid, blk, 0, stream>>>(feats, wb_al, al_b, bufA, CC, 1, xsb, ysb);

    // ---- OR conv (rotated weights, no relu) -> orf ----
    conv_gemm_kernel<0><<<gemm_grid, blk, 0, stream>>>(bufA, wb_or, br_or, orf, CC, 0, nullptr, nullptr);

    // ---- rotation-invariant max pool -> pooled [32][HW] ----
    rot_pool_kernel<<<gcvt(32 * HW), blk, 0, stream>>>(orf, pooled);

    // ---- ODM cls tower (32->256->256) + 3x3 head ----
    conv_gemm_kernel<0><<<gemm_grid, blk, 0, stream>>>(pooled, wb_dc0, dc_b0, bufA, 32, 1, nullptr, nullptr);
    conv_gemm_kernel<0><<<gemm_grid, blk, 0, stream>>>(bufA,   wb_dc1, dc_b1, bufB, CC, 1, nullptr, nullptr);
    head_conv_kernel<<<gcvt(NCLS * HW), blk, 0, stream>>>(bufB, dc_hw, dc_hb, out_odm_cls, CC, NCLS, 3);

    // ---- ODM reg tower (orf -> 256 -> 256) + 3x3 head ----
    conv_gemm_kernel<0><<<gemm_grid, blk, 0, stream>>>(orf,  wb_dr0, dr_b0, bufA, CC, 1, nullptr, nullptr);
    conv_gemm_kernel<0><<<gemm_grid, blk, 0, stream>>>(bufA, wb_dr1, dr_b1, bufB, CC, 1, nullptr, nullptr);
    head_conv_kernel<<<gcvt(5 * HW), blk, 0, stream>>>(bufB, dr_hw, dr_hb, out_odm_bbox, CC, 5, 3);
}
